// EdgeAwareAttention_77214922048083
// MI455X (gfx1250) — compile-verified, matrix operand order checked by reading, not run on hardware
//
#include <hip/hip_runtime.h>
#include <hip/hip_bf16.h>

#ifndef __has_builtin
#define __has_builtin(x) 0
#endif

// Async global->LDS path (CDNA5). Guarded so host pass / older toolchains fall
// back to plain loads while device pass emits global_load_async_to_lds_b32.
#if defined(__HIP_DEVICE_COMPILE__) && __has_builtin(__builtin_amdgcn_global_load_async_to_lds_b32)
#define EA_ASYNC 1
#else
#define EA_ASYNC 0
#endif

typedef __attribute__((ext_vector_type(16))) _Float16 v16h;
typedef __attribute__((ext_vector_type(8)))  float    v8f;

#if EA_ASYNC
typedef __attribute__((address_space(1))) int ea_gint;  // global
typedef __attribute__((address_space(3))) int ea_lint;  // LDS
#endif

__device__ __forceinline__ void ea_async_ld(const float* g, float* l) {
#if EA_ASYNC
  __builtin_amdgcn_global_load_async_to_lds_b32(
      (ea_gint*)g, (ea_lint*)l, /*offset=*/0, /*cpol=*/0);
#else
  *l = *g;
#endif
}

// Wait until at most N async ops outstanding for this wave. In-order
// completion (ISA 10.7/§4.1) makes partial thresholds a pipeline primitive.
__device__ __forceinline__ void ea_async_wait0() {
#if EA_ASYNC
#if __has_builtin(__builtin_amdgcn_s_wait_asynccnt)
  __builtin_amdgcn_s_wait_asynccnt(0);
#else
  asm volatile("s_wait_asynccnt 0" ::: "memory");
#endif
  asm volatile("" ::: "memory");  // keep later LDS stores below the wait
#endif
}
__device__ __forceinline__ void ea_async_wait6() {
#if EA_ASYNC
#if __has_builtin(__builtin_amdgcn_s_wait_asynccnt)
  __builtin_amdgcn_s_wait_asynccnt(6);
#else
  asm volatile("s_wait_asynccnt 6" ::: "memory");
#endif
  asm volatile("" ::: "memory");
#endif
}

// ---------------------------------------------------------------------------
// Problem constants: B=8, C=256, H=W=160, hidden=16, ksize=7
// ---------------------------------------------------------------------------

__global__ void ea_zero_kernel(float* __restrict__ p, int n) {
  int i = blockIdx.x * blockDim.x + threadIdx.x;
  if (i < n) p[i] = 0.f;
}

// ---------------------------------------------------------------------------
// Kernel 1: Sobel + reductions, depth-4 async software pipeline.
// Block = 256 threads = 32x8 pixel tile, loops over 256 channels.
// Every wave issues exactly 2 global_load_async_to_lds_b32 per channel
// (OOB lanes load a clamped valid address into a private scratch slot), so
// s_wait_asynccnt 6 == "channel c landed" while c+1..c+3 stream in.
// ---------------------------------------------------------------------------
__global__ __launch_bounds__(256) void ea_reduce_kernel(const float* __restrict__ x,
                                                        float* __restrict__ csum,
                                                        float* __restrict__ sin_) {
  __shared__ float tile[4][512];  // 10x34=340 used; 512 so every lane owns a slot
  const int tid = threadIdx.x;
  const int tx = tid & 31, ty = tid >> 5;
  const int x0 = blockIdx.x * 32, y0 = blockIdx.y * 8;
  const int b = blockIdx.z;

  float sum_x = 0.f, max_x = -3.402823e38f;
  float sum_g = 0.f, max_g = -3.402823e38f;

  const float* xb = x + (size_t)b * 256 * 25600;

  // Issue loads for channel c into stage buffer: exactly 2 async instrs/wave.
  auto issue = [&](int c, int buf) {
#pragma unroll
    for (int s = 0; s < 2; ++s) {
      const int idx = tid + s * 256;           // private LDS slot (0..511)
      const int id2 = (idx < 340) ? idx : 0;
      const int row = id2 / 34, col = id2 - row * 34;
      const int gy = y0 - 1 + row, gx = x0 - 1 + col;
      const bool ok = (idx < 340) && ((unsigned)gy < 160u) && ((unsigned)gx < 160u);
      const float* xp = xb + (size_t)c * 25600;
      const float* src = ok ? (xp + gy * 160 + gx) : xp;  // clamped, always valid
      ea_async_ld(src, &tile[buf][idx]);
    }
  };
  // Zero the OOB slots this wave owns (safe: runs after this wave's wait).
  auto zfill = [&](int buf) {
#pragma unroll
    for (int s = 0; s < 2; ++s) {
      const int idx = tid + s * 256;
      if (idx < 340) {
        const int row = idx / 34, col = idx - row * 34;
        const int gy = y0 - 1 + row, gx = x0 - 1 + col;
        if (!((unsigned)gy < 160u && (unsigned)gx < 160u)) tile[buf][idx] = 0.f;
      }
    }
  };

  issue(0, 0);
  issue(1, 1);
  issue(2, 2);

  for (int c = 0; c < 256; ++c) {
    if (c + 3 < 256) issue(c + 3, (c + 3) & 3);
    if (c < 253) ea_async_wait6();  // channel c complete; c+1..c+3 in flight
    else         ea_async_wait0();  // drain at the tail
    const float* t = tile[c & 3];
    zfill(c & 3);
    __syncthreads();

    const int r = ty + 1, cc = tx + 1;
    float pmm = t[(r - 1) * 34 + (cc - 1)];
    float pm0 = t[(r - 1) * 34 + cc];
    float pmp = t[(r - 1) * 34 + (cc + 1)];
    float p0m = t[r * 34 + (cc - 1)];
    float p00 = t[r * 34 + cc];
    float p0p = t[r * 34 + (cc + 1)];
    float ppm = t[(r + 1) * 34 + (cc - 1)];
    float pp0 = t[(r + 1) * 34 + cc];
    float ppp = t[(r + 1) * 34 + (cc + 1)];
    float gxs = 0.25f * ((pmm - pmp) + 2.f * (p0m - p0p) + (ppm - ppp));
    float gys = 0.25f * ((pmm + 2.f * pm0 + pmp) - (ppm + 2.f * pp0 + ppp));
    float g = sqrtf(gxs * gxs + gys * gys + 1e-12f);

    sum_x += p00;
    max_x = fmaxf(max_x, p00);
    sum_g += g;
    max_g = fmaxf(max_g, g);

    // wave32 reduction of g for the channel mean
    float w = g;
    for (int off = 16; off > 0; off >>= 1) w += __shfl_down(w, off, 32);
    if ((tid & 31) == 0) atomicAdd(&csum[b * 256 + c], w);

    __syncthreads();  // all waves done with tile[c&3] before it is re-issued
  }

  const size_t pix = (size_t)(y0 + ty) * 160 + (x0 + tx);
  float* sb0 = sin_ + (size_t)b * 4 * 25600;
  sb0[0 * 25600 + pix] = sum_x * (1.f / 256.f);  // avg_x
  sb0[1 * 25600 + pix] = max_x;                  // max_x
  sb0[2 * 25600 + pix] = sum_g * (1.f / 256.f);  // avg_g
  sb0[3 * 25600 + pix] = max_g;                  // max_g
}

// ---------------------------------------------------------------------------
// Kernel 2: channel-gate MLP via WMMA. One wave (32 threads), EXEC all-ones.
// GEMM1: h = relu(c_vec[16pad x 256] @ w1^T[256 x 16])  -> 8x v_wmma_f32_16x16x32_f16
// GEMM2: wgate = sigmoid(h[16 x 32pad] @ w2^T tiles)    -> 16x WMMA
// gmul[b,c] = 1 + softplus(beta) * wgate. Batch padding done with a 0/1 mask
// multiply (no divergent loads; rows 8..15 read owned gmul scratch).
// 16-bit A/B operand layout (ISA 7.12.2): lane -> M (or N) = lane%16;
// VGPR v, lane-half hi: K pair = (v<4 ? 2v : 16+2(v-4)) + (hi?8:0).
// ---------------------------------------------------------------------------
__device__ __forceinline__ int ea_k0(int v, bool hi) {
  return ((v < 4) ? (2 * v) : (16 + 2 * (v - 4))) + (hi ? 8 : 0);
}

__global__ __launch_bounds__(32) void ea_mlp_kernel(const float* __restrict__ csum,
                                                    const float* __restrict__ w1,
                                                    const float* __restrict__ w2,
                                                    const float* __restrict__ beta,
                                                    float* __restrict__ gmul) {
  __shared__ float hbuf[16][16];
  const int lane = threadIdx.x;
  const int M = lane & 15;
  const bool hi = lane >= 16;
  const float amask = (M < 8) ? (1.f / 25600.f) : 0.f;  // c_vec scale * batch pad

  // GEMM1: K = 256 in 8 steps of 32
  v8f acc = {};
#pragma unroll
  for (int kk = 0; kk < 8; ++kk) {
    v16h a, bm;
#pragma unroll
    for (int v = 0; v < 8; ++v) {
      int k0 = ea_k0(v, hi) + kk * 32;
      a[2 * v]     = (_Float16)(csum[M * 256 + k0] * amask);
      a[2 * v + 1] = (_Float16)(csum[M * 256 + k0 + 1] * amask);
      bm[2 * v]     = (_Float16)w1[M * 256 + k0];  // B[k][n] = w1[n][k], n = lane%16
      bm[2 * v + 1] = (_Float16)w1[M * 256 + k0 + 1];
    }
    acc = __builtin_amdgcn_wmma_f32_16x16x32_f16(false, a, false, bm, (short)0, acc,
                                                 false, false);
  }

  // relu, stage h to LDS to re-layout as A operand
#pragma unroll
  for (int r = 0; r < 8; ++r) {
    int m = r + (hi ? 8 : 0);
    hbuf[m][M] = fmaxf(acc[r], 0.f);
  }
  __syncthreads();

  v16h a2;
#pragma unroll
  for (int v = 0; v < 8; ++v) {
    int k0 = ea_k0(v, hi);
    a2[2 * v]     = (k0 < 16) ? (_Float16)hbuf[M][k0] : (_Float16)0.f;
    a2[2 * v + 1] = (k0 + 1 < 16) ? (_Float16)hbuf[M][k0 + 1] : (_Float16)0.f;
  }

  const float bfac = logf(1.f + __expf(beta[0]));  // softplus(beta)
  for (int j = 0; j < 16; ++j) {                   // 16 N-tiles of C=256
    v16h b2;
#pragma unroll
    for (int v = 0; v < 8; ++v) {
      int k0 = ea_k0(v, hi);
      // B2[k][n] = w2[(j*16+n)][k], K padded 16->32 with zeros
      b2[2 * v]     = (k0 < 16) ? (_Float16)w2[(j * 16 + M) * 16 + k0] : (_Float16)0.f;
      b2[2 * v + 1] = (k0 + 1 < 16) ? (_Float16)w2[(j * 16 + M) * 16 + k0 + 1]
                                    : (_Float16)0.f;
    }
    v8f acc2 = {};
    acc2 = __builtin_amdgcn_wmma_f32_16x16x32_f16(false, a2, false, b2, (short)0, acc2,
                                                  false, false);
#pragma unroll
    for (int r = 0; r < 8; ++r) {
      int m = r + (hi ? 8 : 0);
      if (m < 8) {
        float sg = 1.f / (1.f + __expf(-acc2[r]));
        gmul[m * 256 + j * 16 + M] = 1.f + bfac * sg;
      }
    }
  }
}

// ---------------------------------------------------------------------------
// Kernel 3: 7x7 spatial conv over the 4-channel descriptor + final apply.
// Block = 32x8 tile. Stages s_in tile (+3 halo) into LDS with exactly 9
// async instrs/wave (clamped OOB, private slots), computes
// smul = 1 + softplus(alpha)*sigmoid(conv+bias), then streams all 256
// channels: out = x * smul * gmul[b,c] (coalesced 128B rows).
// ---------------------------------------------------------------------------
__global__ __launch_bounds__(256) void ea_apply_kernel(const float* __restrict__ x,
                                                       const float* __restrict__ sin_,
                                                       const float* __restrict__ gmul,
                                                       const float* __restrict__ sw,
                                                       const float* __restrict__ sbias,
                                                       const float* __restrict__ alpha,
                                                       float* __restrict__ out) {
  __shared__ float st[2304];  // 4 ch x 14 x 38 = 2128 used; 9*256 slots
  __shared__ float wg[196];
  __shared__ float gm[256];
  const int tid = threadIdx.x;
  const int tx = tid & 31, ty = tid >> 5;
  const int x0 = blockIdx.x * 32, y0 = blockIdx.y * 8;
  const int b = blockIdx.z;

  const float* sb0 = sin_ + (size_t)b * 4 * 25600;
#pragma unroll
  for (int s = 0; s < 9; ++s) {
    const int idx = tid + s * 256;          // private LDS slot
    const int id2 = (idx < 2128) ? idx : 0;
    const int f = id2 / 532;
    const int e = id2 - f * 532;
    const int row = e / 38, col = e - row * 38;
    const int gy = y0 - 3 + row, gx = x0 - 3 + col;
    const bool ok = (idx < 2128) && ((unsigned)gy < 160u) && ((unsigned)gx < 160u);
    const float* src = ok ? (sb0 + f * 25600 + gy * 160 + gx) : sb0;
    ea_async_ld(src, &st[idx]);
  }
  if (tid < 196) wg[tid] = sw[tid];
  gm[tid] = gmul[b * 256 + tid];

  ea_async_wait0();
  // zero-fill OOB slots this wave owns (after own wait -> race-free)
#pragma unroll
  for (int s = 0; s < 9; ++s) {
    const int idx = tid + s * 256;
    if (idx < 2128) {
      const int f = idx / 532;
      const int e = idx - f * 532;
      const int row = e / 38, col = e - row * 38;
      const int gy = y0 - 3 + row, gx = x0 - 3 + col;
      if (!((unsigned)gy < 160u && (unsigned)gx < 160u)) st[idx] = 0.f;
    }
  }
  __syncthreads();

  float acc = sbias[0];
#pragma unroll
  for (int ch = 0; ch < 4; ++ch)
    for (int ky = 0; ky < 7; ++ky)
#pragma unroll
      for (int kx = 0; kx < 7; ++kx)
        acc += st[ch * 532 + (ty + ky) * 38 + (tx + kx)] * wg[ch * 49 + ky * 7 + kx];

  const float afac = logf(1.f + __expf(alpha[0]));  // softplus(alpha)
  const float smul = 1.f + afac * (1.f / (1.f + __expf(-acc)));

  const size_t base = (size_t)b * 256 * 25600 + (size_t)(y0 + ty) * 160 + (x0 + tx);
  const float* xp = x + base;
  float* op = out + base;
#pragma unroll 4
  for (int c = 0; c < 256; ++c) {
    op[(size_t)c * 25600] = xp[(size_t)c * 25600] * (smul * gm[c]);
  }
}

// ---------------------------------------------------------------------------
extern "C" void kernel_launch(void* const* d_in, const int* in_sizes, int n_in,
                              void* d_out, int out_size, void* d_ws, size_t ws_size,
                              hipStream_t stream) {
  (void)in_sizes; (void)n_in; (void)out_size; (void)ws_size;
  const float* x     = (const float*)d_in[0];
  const float* w1    = (const float*)d_in[1];
  const float* w2    = (const float*)d_in[2];
  const float* sw    = (const float*)d_in[3];
  const float* sb    = (const float*)d_in[4];
  const float* alpha = (const float*)d_in[5];
  const float* beta  = (const float*)d_in[6];
  float* out = (float*)d_out;

  float* ws   = (float*)d_ws;
  float* csum = ws;            // [8*256]
  float* gmul = ws + 2048;     // [8*256]
  float* sin_ = ws + 4096;     // [8*4*160*160]

  ea_zero_kernel<<<8, 256, 0, stream>>>(csum, 2048);

  dim3 grid(5, 20, 8);  // 160/32 x 160/8 x B
  ea_reduce_kernel<<<grid, 256, 0, stream>>>(x, csum, sin_);
  ea_mlp_kernel<<<1, 32, 0, stream>>>(csum, w1, w2, beta, gmul);
  ea_apply_kernel<<<grid, 256, 0, stream>>>(x, sin_, gmul, sw, sb, alpha, out);
}
